// PointNext_13374528159876
// MI455X (gfx1250) — matrix operand order repaired and checked by reading, not compiled, not run on hardware
//
#include <hip/hip_runtime.h>

// PointNext LocalAggregation for MI455X (gfx1250, wave32, WMMA).
//
// Pipeline (all on `stream`, graph-capture safe, deterministic: no atomics):
//   1) knn_kernel:      brute-force top-16 per query, register-resident sorted list
//   2) mlp_pass<0>:     gather -> GEMM1(WMMA f16) -> per-block BN1 partial stats
//   3) finalize_stats:  deterministic reduction -> scale1/shift1
//   4) mlp_pass<1>:     gather -> GEMM1 -> BN1+ReLU -> GEMM2 -> BN2 partial stats
//   5) finalize_stats:  -> scale2/shift2
//   6) mlp_pass<2>:     full recompute -> BN2+ReLU -> max over K -> out
//
// Roofline: feat (16MB) + coord (768KB) + weights live in the 192MB L2, so the
// recompute strategy replaces ~1GB of HBM intermediate traffic with cheap
// L2-fed WMMA work (GEMMs are only ~17 GFLOP total).

typedef _Float16 half_t;
typedef __attribute__((ext_vector_type(16))) _Float16 v16h;
typedef __attribute__((ext_vector_type(8)))  _Float16 v8h;
typedef __attribute__((ext_vector_type(8)))  float    v8f;

#define KQ    16      // k neighbors
#define ROWS  128     // batch-rows per workgroup tile (= 8 points * 16 neighbors)
#define LDH   72      // LDS half-stride for 64-wide activation rows (144B, 16B-aligned)
#define LDW   72      // LDS half-stride for transposed weight columns
#define LDO   65      // LDS float-stride for the output tile
#define CEPS  1e-5f

// ---------------------------------------------------------------------------
// helpers
// ---------------------------------------------------------------------------
__device__ __forceinline__ v16h load2x8(const half_t* p0, const half_t* p1) {
  // two 16-byte contiguous chunks -> ds_load_b128 x2
  v8h lo = *(const v8h*)p0;
  v8h hi = *(const v8h*)p1;
  v16h r;
#pragma unroll
  for (int e = 0; e < 8; ++e) { r[e] = lo[e]; r[e + 8] = hi[e]; }
  return r;
}

__device__ __forceinline__ v8f wmma16(v16h a, v16h b, v8f c) {
  return __builtin_amdgcn_wmma_f32_16x16x32_f16(false, a, false, b, (short)0, c,
                                                false, false);
}

// ---------------------------------------------------------------------------
// 1) Brute-force exact KNN (top-16, ascending, ties keep lower index)
// ---------------------------------------------------------------------------
__global__ __launch_bounds__(256) void knn_kernel(const float* __restrict__ coord,
                                                  int n,
                                                  int* __restrict__ idx_out) {
  __shared__ float sx[256], sy[256], sz[256], ssq[256];
  const int tid = threadIdx.x;
  const int q = blockIdx.x * 256 + tid;
  const float qx = coord[q * 3 + 0];
  const float qy = coord[q * 3 + 1];
  const float qz = coord[q * 3 + 2];

  float bd[16];
  int   bi[16];
#pragma unroll
  for (int j = 0; j < 16; ++j) { bd[j] = 3.0e38f; bi[j] = 0; }

  for (int base = 0; base < n; base += 256) {
    const int c = base + tid;
    const float cx = coord[c * 3 + 0];
    const float cy = coord[c * 3 + 1];
    const float cz = coord[c * 3 + 2];
    sx[tid] = cx; sy[tid] = cy; sz[tid] = cz;
    ssq[tid] = cx * cx + cy * cy + cz * cz;
    __syncthreads();
#pragma unroll 4
    for (int t = 0; t < 256; ++t) {
      // |c|^2 - 2 q.c  ==  |q-c|^2 - |q|^2  (constant shift per query: same ranking)
      const float d = ssq[t] - 2.0f * (qx * sx[t] + qy * sy[t] + qz * sz[t]);
      if (d < bd[15]) {
        float cd = d; int ci = base + t;
#pragma unroll
        for (int j = 0; j < 16; ++j) {
          if (cd < bd[j]) {
            float td = bd[j]; int ti = bi[j];
            bd[j] = cd; bi[j] = ci;
            cd = td; ci = ti;
          }
        }
      }
    }
    __syncthreads();
  }
#pragma unroll
  for (int j = 0; j < 16; ++j) idx_out[(size_t)q * 16 + j] = bi[j];
}

// ---------------------------------------------------------------------------
// 2/4/6) Fused gather + MLP pass.  PHASE 0: BN1 stats. PHASE 1: BN2 stats.
//        PHASE 2: full forward + max over K.
// LDS arena (bytes):
//   0      sA   [128][72] f16  (gather tile)        18432
//   18432  sW1t [64][72]  f16  (W1 transposed)       9216
//   27648  sNb  [128]     i32                         512
//   28160  sRedS[1024]    f32  (stat partials)       4096
//   32256  sRedQ[1024]    f32                        4096
//   36352  sW2t [64][72]  f16  (PHASE>=1)            9216
//   45568  sH1  [128][72] f16  (PHASE>=1)           18432   -> 64000 total
//   0      sOut [128][65] f32  (PHASE 2, aliases sA/sW1t region, post-barrier)
// ---------------------------------------------------------------------------
template <int PHASE>
__global__ __launch_bounds__(256) void mlp_pass(
    const float* __restrict__ feat, const int* __restrict__ knn,
    const float* __restrict__ W1, const float* __restrict__ b1g,
    const float* __restrict__ W2, const float* __restrict__ b2g,
    const float* __restrict__ scale1, const float* __restrict__ shift1,
    const float* __restrict__ scale2, const float* __restrict__ shift2,
    float* __restrict__ partial, float* __restrict__ out) {
  constexpr int SMEM_BYTES = (PHASE == 0) ? 36352 : 64000;
  __shared__ __align__(16) unsigned char smem[SMEM_BYTES];
  half_t* sA    = (half_t*)(smem + 0);
  half_t* sW1t  = (half_t*)(smem + 18432);
  int*    sNb   = (int*)   (smem + 27648);
  float*  sRedS = (float*) (smem + 28160);
  float*  sRedQ = (float*) (smem + 32256);
  half_t* sW2t  = (half_t*)(smem + 36352);  // valid only for PHASE>=1
  half_t* sH1   = (half_t*)(smem + 45568);  // valid only for PHASE>=1
  float*  sOut  = (float*) (smem + 0);      // PHASE==2 only (aliases gather tile)

  const int tid = threadIdx.x;
  const size_t rowbase = (size_t)blockIdx.x * ROWS;

  // ---- stage: neighbor ids + transposed weights (f32 -> f16) ----
  if (tid < ROWS) sNb[tid] = knn[rowbase + tid];
  for (int e = tid; e < 64 * 64; e += 256) {
    const int k = e >> 6, nn = e & 63;
    sW1t[nn * LDW + k] = (half_t)W1[e];
    if (PHASE >= 1) sW2t[nn * LDW + k] = (half_t)W2[e];
  }
  __syncthreads();

  // ---- stage: gather neighbor feature rows (L2-resident), f32 -> f16 ----
  {
    const float4* f4 = (const float4*)feat;
    for (int e = tid; e < ROWS * 16; e += 256) {
      const int r = e >> 4, c4 = e & 15;
      const float4 v = f4[(size_t)sNb[r] * 16 + c4];
      half_t* dst = sA + r * LDH + c4 * 4;
      dst[0] = (half_t)v.x; dst[1] = (half_t)v.y;
      dst[2] = (half_t)v.z; dst[3] = (half_t)v.w;
    }
  }
  __syncthreads();

  const int wave  = tid >> 5;
  const int lane  = tid & 31;
  const int laneM = lane & 15;   // A row / B column within tile; C channel lane
  const int hi    = lane >> 4;

  // ================= GEMM1: grouped[128x64] @ W1 =================
  {
    const half_t* aRow = sA + (wave * 16 + laneM) * LDH;
    const v16h a0 = load2x8(aRow + hi * 8,      aRow + hi * 8 + 16);   // K 0..31
    const v16h a1 = load2x8(aRow + 32 + hi * 8, aRow + 48 + hi * 8);   // K 32..63
#pragma unroll
    for (int nt = 0; nt < 4; ++nt) {
      const int ch = nt * 16 + laneM;
      const half_t* wcol = sW1t + ch * LDW;
      const v16h bq0 = load2x8(wcol + hi * 8,      wcol + hi * 8 + 16);
      const v16h bq1 = load2x8(wcol + 32 + hi * 8, wcol + 48 + hi * 8);
      v8f acc = {};
      acc = wmma16(a0, bq0, acc);
      acc = wmma16(a1, bq1, acc);
      const float bias = b1g[ch];
      if constexpr (PHASE == 0) {
        float s = 0.f, q = 0.f;
#pragma unroll
        for (int v = 0; v < 8; ++v) { const float h = acc[v] + bias; s += h; q += h * h; }
        sRedS[wave * 128 + hi * 64 + ch] = s;
        sRedQ[wave * 128 + hi * 64 + ch] = q;
      } else {
        const float sc = scale1[ch], sh = shift1[ch];
#pragma unroll
        for (int v = 0; v < 8; ++v) {
          const float h = fmaxf(fmaf(acc[v] + bias, sc, sh), 0.f);
          sH1[(wave * 16 + v + hi * 8) * LDH + ch] = (half_t)h;
        }
      }
    }
  }

  // ================= GEMM2: h1[128x64] @ W2 (PHASE >= 1) =================
  if constexpr (PHASE >= 1) {
    __syncthreads();
    const half_t* hRow = sH1 + (wave * 16 + laneM) * LDH;
    const v16h a0 = load2x8(hRow + hi * 8,      hRow + hi * 8 + 16);
    const v16h a1 = load2x8(hRow + 32 + hi * 8, hRow + 48 + hi * 8);
#pragma unroll
    for (int nt = 0; nt < 4; ++nt) {
      const int ch = nt * 16 + laneM;
      const half_t* wcol = sW2t + ch * LDW;
      const v16h bq0 = load2x8(wcol + hi * 8,      wcol + hi * 8 + 16);
      const v16h bq1 = load2x8(wcol + 32 + hi * 8, wcol + 48 + hi * 8);
      v8f acc = {};
      acc = wmma16(a0, bq0, acc);
      acc = wmma16(a1, bq1, acc);
      const float bias = b2g[ch];
      if constexpr (PHASE == 1) {
        float s = 0.f, q = 0.f;
#pragma unroll
        for (int v = 0; v < 8; ++v) { const float h = acc[v] + bias; s += h; q += h * h; }
        sRedS[wave * 128 + hi * 64 + ch] = s;
        sRedQ[wave * 128 + hi * 64 + ch] = q;
      } else {
        const float sc = scale2[ch], sh = shift2[ch];
#pragma unroll
        for (int v = 0; v < 8; ++v) {
          const float h = fmaxf(fmaf(acc[v] + bias, sc, sh), 0.f);
          sOut[(wave * 16 + v + hi * 8) * LDO + ch] = h;
        }
      }
    }
  }

  // ---- per-block partial stats (fixed-order -> deterministic) ----
  if constexpr (PHASE == 0 || PHASE == 1) {
    __syncthreads();
    if (tid < 64) {
      float s = 0.f, q = 0.f;
#pragma unroll
      for (int w = 0; w < 8; ++w) {
        s += sRedS[w * 128 + tid] + sRedS[w * 128 + 64 + tid];
        q += sRedQ[w * 128 + tid] + sRedQ[w * 128 + 64 + tid];
      }
      partial[(size_t)blockIdx.x * 128 + tid]      = s;
      partial[(size_t)blockIdx.x * 128 + 64 + tid] = q;
    }
  }

  // ---- max over K neighbors -> output (PHASE 2) ----
  if constexpr (PHASE == 2) {
    __syncthreads();
    const size_t pbase = rowbase / KQ;  // 8 points per block
    for (int o = tid; o < (ROWS / KQ) * 64; o += 256) {
      const int p = o >> 6, c = o & 63;
      const float* col = sOut + (p * 16) * LDO + c;
      float m = col[0];
#pragma unroll
      for (int r = 1; r < 16; ++r) m = fmaxf(m, col[r * LDO]);
      out[(pbase + p) * 64 + c] = m;
    }
  }
}

// ---------------------------------------------------------------------------
// 3/5) Deterministic finalize: partials -> BN scale/shift
// ---------------------------------------------------------------------------
__global__ __launch_bounds__(64) void finalize_stats(const float* __restrict__ partial,
                                                     int nblocks, float invN,
                                                     const float* __restrict__ gamma,
                                                     const float* __restrict__ beta,
                                                     float* __restrict__ scale,
                                                     float* __restrict__ shift) {
  const int c = threadIdx.x;  // 64 channels
  float s = 0.f, q = 0.f;
  for (int b = 0; b < nblocks; ++b) {
    s += partial[(size_t)b * 128 + c];
    q += partial[(size_t)b * 128 + 64 + c];
  }
  const float mean = s * invN;
  const float var  = q * invN - mean * mean;
  const float sc   = gamma[c] * rsqrtf(var + CEPS);
  scale[c] = sc;
  shift[c] = beta[c] - mean * sc;
}

// ---------------------------------------------------------------------------
// launcher
// ---------------------------------------------------------------------------
extern "C" void kernel_launch(void* const* d_in, const int* in_sizes, int n_in,
                              void* d_out, int out_size, void* d_ws, size_t ws_size,
                              hipStream_t stream) {
  const float* coord = (const float*)d_in[0];
  const float* feat  = (const float*)d_in[1];
  // d_in[2] = offset scalar (unused)
  const float* W1  = (const float*)d_in[3];
  const float* b1  = (const float*)d_in[4];
  const float* g1  = (const float*)d_in[5];
  const float* be1 = (const float*)d_in[6];
  const float* W2  = (const float*)d_in[7];
  const float* b2  = (const float*)d_in[8];
  const float* g2  = (const float*)d_in[9];
  const float* be2 = (const float*)d_in[10];
  float* out = (float*)d_out;

  const int n      = in_sizes[0] / 3;   // 65536
  const int nrows  = n * KQ;            // 1048576
  const int nblk   = nrows / ROWS;      // 8192
  const float invN = 1.0f / (float)nrows;

  // workspace carve
  char* ws = (char*)d_ws;
  int*   knnIdx  = (int*)ws;                                   // n*16 ints   (4 MB)
  float* partial = (float*)(ws + (size_t)n * KQ * 4);          // nblk*128 f32 (4 MB)
  float* sc      = (float*)(ws + (size_t)n * KQ * 4 + (size_t)nblk * 128 * 4);
  float* scale1 = sc, * shift1 = sc + 64, * scale2 = sc + 128, * shift2 = sc + 192;

  knn_kernel<<<n / 256, 256, 0, stream>>>(coord, n, knnIdx);

  mlp_pass<0><<<nblk, 256, 0, stream>>>(feat, knnIdx, W1, b1, W2, b2,
                                        scale1, shift1, scale2, shift2, partial, out);
  finalize_stats<<<1, 64, 0, stream>>>(partial, nblk, invN, g1, be1, scale1, shift1);

  mlp_pass<1><<<nblk, 256, 0, stream>>>(feat, knnIdx, W1, b1, W2, b2,
                                        scale1, shift1, scale2, shift2, partial, out);
  finalize_stats<<<1, 64, 0, stream>>>(partial, nblk, invN, g2, be2, scale2, shift2);

  mlp_pass<2><<<nblk, 256, 0, stream>>>(feat, knnIdx, W1, b1, W2, b2,
                                        scale1, shift1, scale2, shift2, partial, out);
}
